// FusedMoEModel_36833639531197
// MI455X (gfx1250) — compile-verified, matrix operand order checked by reading, not run on hardware
//
#include <hip/hip_runtime.h>
#include <hip/hip_bf16.h>
#include <math.h>

#define TOKENS   4096
#define HID      1024
#define INTER    2048
#define NEXP     8
#define ROWS     (TOKENS * 2)   /* exactly TOKENS*TOP_K gathered rows */
#define W1ROWS   (2 * INTER)    /* 4096 */
#define KH1      (HID / 2)      /* 512 packed int32 per w1 row  */
#define KH2      (INTER / 2)    /* 1024 packed int32 per w2 row */

typedef __bf16          v16bf  __attribute__((ext_vector_type(16)));
typedef float           v8f    __attribute__((ext_vector_type(8)));
typedef unsigned short  u16x8  __attribute__((ext_vector_type(8)));
typedef unsigned short  u16x4  __attribute__((ext_vector_type(4)));
typedef int             i32x4  __attribute__((ext_vector_type(4)));

union FragBF {
    v16bf v;
    u16x8 h[2];
};

/* ---------- CDNA5 async global->LDS path (ASYNCcnt), with safe fallback ---------- */
#define AS1 __attribute__((address_space(1)))
#define AS3 __attribute__((address_space(3)))

#if __has_builtin(__builtin_amdgcn_global_load_async_to_lds_b128)
#define HAVE_ASYNC_LDS 1
__device__ __forceinline__ void async_g2l_b128(void* lds, const void* gsrc) {
    __builtin_amdgcn_global_load_async_to_lds_b128(
        (AS1 i32x4*)(gsrc), (AS3 i32x4*)(lds), 0, 0);
}
#else
#define HAVE_ASYNC_LDS 0
#endif

#if __has_builtin(__builtin_amdgcn_s_wait_asynccnt)
#define ASYNC_WAIT() __builtin_amdgcn_s_wait_asynccnt(0)
#else
#define ASYNC_WAIT() asm volatile("s_wait_asynccnt 0" ::: "memory")
#endif

__device__ __forceinline__ unsigned short f2bf(float f) {
    unsigned u = __builtin_bit_cast(unsigned, f);
    u += 0x7fffu + ((u >> 16) & 1u);          // round-to-nearest-even
    return (unsigned short)(u >> 16);
}

// unpack 4 packed int32 (one byte each = 2 nibbles) -> 8 bf16 weights into LDS
__device__ __forceinline__ void dequant8(int4 p, float s, unsigned short* dst) {
    int v[4] = { p.x, p.y, p.z, p.w };
    u16x8 o;
#pragma unroll
    for (int j = 0; j < 4; j++) {
        float f0 = (float)(( v[j]       & 15) - 8) * s;
        float f1 = (float)(((v[j] >> 4) & 15) - 8) * s;
        o[2 * j]     = f2bf(f0);
        o[2 * j + 1] = f2bf(f1);
    }
    *(u16x8*)dst = o;
}

/* ---------------- zero output ---------------- */
__global__ void moe_zero(float4* out) {
    out[blockIdx.x * blockDim.x + threadIdx.x] = float4{0.f, 0.f, 0.f, 0.f};
}

/* ---------------- x -> bf16 ---------------- */
__global__ void moe_cvt(const float4* __restrict__ x, u16x4* __restrict__ xb) {
    int i = blockIdx.x * blockDim.x + threadIdx.x;
    float4 v = x[i];
    u16x4 o;
    o[0] = f2bf(v.x); o[1] = f2bf(v.y); o[2] = f2bf(v.z); o[3] = f2bf(v.w);
    xb[i] = o;
}

/* ---------------- router: scores, top-2, softmax ---------------- */
__global__ __launch_bounds__(256) void moe_router(
    const float* __restrict__ x, const float* __restrict__ rw,
    int* __restrict__ top_i, float* __restrict__ top_w)
{
    __shared__ float rws[NEXP * HID];               // 32 KB
    int tid = threadIdx.x;
    for (int i = tid; i < NEXP * HID; i += 256) rws[i] = rw[i];
    __syncthreads();

    int t = blockIdx.x * 256 + tid;
    float s[NEXP];
#pragma unroll
    for (int e = 0; e < NEXP; e++) s[e] = 0.f;

    const float4* xr = (const float4*)(x + (size_t)t * HID);
    for (int k4 = 0; k4 < HID / 4; k4++) {
        float4 xv = xr[k4];
#pragma unroll
        for (int e = 0; e < NEXP; e++) {
            const float* re = &rws[e * HID + k4 * 4];
            s[e] += xv.x * re[0] + xv.y * re[1] + xv.z * re[2] + xv.w * re[3];
        }
    }
    // top-2 (ties -> lowest index, like jax top_k)
    int i0 = 0; float m0 = s[0];
#pragma unroll
    for (int e = 1; e < NEXP; e++) if (s[e] > m0) { m0 = s[e]; i0 = e; }
    int i1 = -1; float m1 = -3.4e38f;
#pragma unroll
    for (int e = 0; e < NEXP; e++) if (e != i0 && s[e] > m1) { m1 = s[e]; i1 = e; }
    float w0 = 1.f / (1.f + __expf(m1 - m0));       // softmax over [m0,m1]
    top_i[2 * t] = i0;  top_i[2 * t + 1] = i1;
    top_w[2 * t] = w0;  top_w[2 * t + 1] = 1.f - w0;
}

/* ------------- deterministic per-expert compaction ------------- */
__global__ void moe_compact(const int* __restrict__ top_i, const float* __restrict__ top_w,
                            int* __restrict__ expert_off, int* __restrict__ tok_idx,
                            float* __restrict__ ge_row)
{
    __shared__ int cnt[NEXP];
    __shared__ int off[NEXP + 1];
    int tid = threadIdx.x;
    if (tid < NEXP) {
        int c = 0;
        for (int t = 0; t < TOKENS; t++) {
            int a = top_i[2 * t], b = top_i[2 * t + 1];
            c += (a == tid) + (b == tid);
        }
        cnt[tid] = c;
    }
    __syncthreads();
    if (tid == 0) {
        int o = 0;
        for (int e = 0; e < NEXP; e++) { off[e] = o; o += cnt[e]; }
        off[NEXP] = o;
        for (int e = 0; e <= NEXP; e++) expert_off[e] = off[e];
    }
    __syncthreads();
    if (tid < NEXP) {
        int p = off[tid];
        for (int t = 0; t < TOKENS; t++) {
            int a = top_i[2 * t], b = top_i[2 * t + 1];
            if (a == tid) { tok_idx[p] = t; ge_row[p] = top_w[2 * t];     p++; }
            if (b == tid) { tok_idx[p] = t; ge_row[p] = top_w[2 * t + 1]; p++; }
        }
    }
}

/* ------------- GEMM1: a = silu(x W1g^T) * (x W1u^T), gathered rows ------------- */
__global__ __launch_bounds__(256) void moe_gemm1(
    const unsigned short* __restrict__ xb,
    const int* __restrict__ w1, const float* __restrict__ w1s,
    const int* __restrict__ expert_off, const int* __restrict__ tok_idx,
    unsigned short* __restrict__ a_out)
{
    __shared__ __align__(16) unsigned short As[64][40];
    __shared__ __align__(16) unsigned short Bg[64][40];
    __shared__ __align__(16) unsigned short Bu[64][40];
    __shared__ float Gs[64][68];

    const int e  = blockIdx.z;
    const int mt = blockIdx.y;
    const int n0 = blockIdx.x * 64;
    const int base  = expert_off[e];
    const int count = expert_off[e + 1] - base;
    if (mt * 64 >= count) return;
    const int row0 = base + mt * 64;

    const int tid  = threadIdx.x;
    const int lane = tid & 31;
    const int wv   = tid >> 5;
    const int lr   = lane & 15;
    const int lh   = lane >> 4;
    const int msub = wv & 3;
    const int half = wv >> 2;

    // staging mapping: 64 rows x 4 chunks of 8 k
    const int  sr = tid >> 2;
    const int  sc = tid & 3;
    const bool rowok = (mt * 64 + sr) < count;
    const int  tok = rowok ? tok_idx[row0 + sr] : 0;
    const unsigned short* aptr = xb + (size_t)tok * HID + sc * 8;
    const int*   gptr = w1  + ((size_t)e * W1ROWS + n0 + sr) * KH1 + sc * 4;
    const int*   uptr = w1  + ((size_t)e * W1ROWS + INTER + n0 + sr) * KH1 + sc * 4;
    const float* gsp  = w1s + ((size_t)e * W1ROWS + n0 + sr) * (HID / 128);
    const float* usp  = w1s + ((size_t)e * W1ROWS + INTER + n0 + sr) * (HID / 128);

    v8f acc[4];
#pragma unroll
    for (int i = 0; i < 4; i++) acc[i] = v8f{};

#if HAVE_ASYNC_LDS
    // slots for out-of-range rows are never touched by the async engine:
    // zero them exactly once before the K loop.
    if (!rowok) *(u16x8*)&As[sr][sc * 8] = u16x8{};
#endif

    for (int kb = 0; kb < HID; kb += 32) {
#if HAVE_ASYNC_LDS
        if (rowok) async_g2l_b128(&As[sr][sc * 8], aptr + kb);
#else
        u16x8 av = u16x8{};
        if (rowok) av = *(const u16x8*)(aptr + kb);
        *(u16x8*)&As[sr][sc * 8] = av;
#endif
        dequant8(*(const int4*)(gptr + (kb >> 1)), gsp[kb >> 7], &Bg[sr][sc * 8]);
        dequant8(*(const int4*)(uptr + (kb >> 1)), usp[kb >> 7], &Bu[sr][sc * 8]);
        if (kb + 32 < HID) {
            __builtin_prefetch(gptr + ((kb + 32) >> 1), 0, 1);
            __builtin_prefetch(uptr + ((kb + 32) >> 1), 0, 1);
        }
#if HAVE_ASYNC_LDS
        ASYNC_WAIT();
#endif
        __syncthreads();

        FragBF a;
        a.h[0] = *(const u16x8*)&As[msub * 16 + lr][lh * 8];
        a.h[1] = *(const u16x8*)&As[msub * 16 + lr][16 + lh * 8];
        const unsigned short (*Bt)[40] = half ? Bu : Bg;
#pragma unroll
        for (int ns = 0; ns < 4; ns++) {
            FragBF b;
            b.h[0] = *(const u16x8*)&Bt[ns * 16 + lr][lh * 8];
            b.h[1] = *(const u16x8*)&Bt[ns * 16 + lr][16 + lh * 8];
            acc[ns] = __builtin_amdgcn_wmma_f32_16x16x32_bf16(
                false, a.v, false, b.v, (short)0, acc[ns], false, false);
        }
        __syncthreads();
    }

    if (half == 0) {
#pragma unroll
        for (int ns = 0; ns < 4; ns++)
#pragma unroll
            for (int j = 0; j < 8; j++)
                Gs[msub * 16 + j + 8 * lh][ns * 16 + lr] = acc[ns][j];
    }
    __syncthreads();
    if (half == 1) {
#pragma unroll
        for (int ns = 0; ns < 4; ns++)
#pragma unroll
            for (int j = 0; j < 8; j++) {
                int rl = msub * 16 + j + 8 * lh;
                if (mt * 64 + rl < count) {
                    float g = Gs[rl][ns * 16 + lr];
                    float u = acc[ns][j];
                    float av2 = g * (1.f / (1.f + __expf(-g))) * u;
                    a_out[(size_t)(row0 + rl) * INTER + n0 + ns * 16 + lr] = f2bf(av2);
                }
            }
    }
}

/* ------------- GEMM2: out[tok] += ge * (a W2^T) ------------- */
__global__ __launch_bounds__(256) void moe_gemm2(
    const unsigned short* __restrict__ a_bf,
    const int* __restrict__ w2, const float* __restrict__ w2s,
    const int* __restrict__ expert_off, const int* __restrict__ tok_idx,
    const float* __restrict__ ge_row,
    float* __restrict__ out)
{
    __shared__ __align__(16) unsigned short As[64][40];
    __shared__ __align__(16) unsigned short Bs[64][40];

    const int e  = blockIdx.z;
    const int mt = blockIdx.y;
    const int n0 = blockIdx.x * 64;
    const int base  = expert_off[e];
    const int count = expert_off[e + 1] - base;
    if (mt * 64 >= count) return;
    const int row0 = base + mt * 64;

    const int tid  = threadIdx.x;
    const int lane = tid & 31;
    const int wv   = tid >> 5;
    const int lr   = lane & 15;
    const int lh   = lane >> 4;
    const int msub = wv & 3;
    const int nh   = wv >> 2;

    const int  sr = tid >> 2;
    const int  sc = tid & 3;
    const bool rowok = (mt * 64 + sr) < count;
    const unsigned short* aptr = a_bf + (size_t)(row0 + sr) * INTER + sc * 8;
    const int*   bptr = w2  + ((size_t)e * HID + n0 + sr) * KH2 + sc * 4;
    const float* bsp  = w2s + ((size_t)e * HID + n0 + sr) * (INTER / 128);

    v8f acc[2];
    acc[0] = v8f{}; acc[1] = v8f{};

#if HAVE_ASYNC_LDS
    if (!rowok) *(u16x8*)&As[sr][sc * 8] = u16x8{};
#endif

    for (int kb = 0; kb < INTER; kb += 32) {
#if HAVE_ASYNC_LDS
        if (rowok) async_g2l_b128(&As[sr][sc * 8], aptr + kb);
#else
        u16x8 av = u16x8{};
        if (rowok) av = *(const u16x8*)(aptr + kb);
        *(u16x8*)&As[sr][sc * 8] = av;
#endif
        dequant8(*(const int4*)(bptr + (kb >> 1)), bsp[kb >> 7], &Bs[sr][sc * 8]);
        if (kb + 32 < INTER) __builtin_prefetch(bptr + ((kb + 32) >> 1), 0, 1);
#if HAVE_ASYNC_LDS
        ASYNC_WAIT();
#endif
        __syncthreads();

        FragBF a;
        a.h[0] = *(const u16x8*)&As[msub * 16 + lr][lh * 8];
        a.h[1] = *(const u16x8*)&As[msub * 16 + lr][16 + lh * 8];
#pragma unroll
        for (int ni = 0; ni < 2; ni++) {
            FragBF b;
            b.h[0] = *(const u16x8*)&Bs[nh * 32 + ni * 16 + lr][lh * 8];
            b.h[1] = *(const u16x8*)&Bs[nh * 32 + ni * 16 + lr][16 + lh * 8];
            acc[ni] = __builtin_amdgcn_wmma_f32_16x16x32_bf16(
                false, a.v, false, b.v, (short)0, acc[ni], false, false);
        }
        __syncthreads();
    }

    // exactly 2 commutative float atomic adds per output element -> deterministic
#pragma unroll
    for (int ni = 0; ni < 2; ni++)
#pragma unroll
        for (int j = 0; j < 8; j++) {
            int rl = msub * 16 + j + 8 * lh;
            if (mt * 64 + rl < count) {
                int row = row0 + rl;
                float g = ge_row[row];
                int tok = tok_idx[row];
                atomicAdd(out + (size_t)tok * HID + n0 + nh * 32 + ni * 16 + lr,
                          g * acc[ni][j]);
            }
        }
}

/* ---------------- launcher ---------------- */
extern "C" void kernel_launch(void* const* d_in, const int* in_sizes, int n_in,
                              void* d_out, int out_size, void* d_ws, size_t ws_size,
                              hipStream_t stream) {
    const float* x    = (const float*)d_in[0];
    const float* rw   = (const float*)d_in[1];
    const int*   w1   = (const int*)d_in[2];
    const float* w1s  = (const float*)d_in[3];
    const int*   w2   = (const int*)d_in[4];
    const float* w2s  = (const float*)d_in[5];
    float* out = (float*)d_out;

    char* ws = (char*)d_ws;
    size_t off = 0;
    auto walloc = [&](size_t bytes) -> char* {
        char* p = ws + off;
        off = (off + bytes + 255) & ~(size_t)255;
        return p;
    };
    int*   top_i      = (int*)  walloc((size_t)TOKENS * 2 * 4);
    float* top_w      = (float*)walloc((size_t)TOKENS * 2 * 4);
    int*   expert_off = (int*)  walloc(16 * 4);
    int*   tok_idx    = (int*)  walloc((size_t)ROWS * 4);
    float* ge_row     = (float*)walloc((size_t)ROWS * 4);
    unsigned short* xb   = (unsigned short*)walloc((size_t)TOKENS * HID * 2);
    unsigned short* a_bf = (unsigned short*)walloc((size_t)ROWS * INTER * 2);

    moe_zero<<<(TOKENS * HID) / (256 * 4), 256, 0, stream>>>((float4*)out);
    moe_router<<<TOKENS / 256, 256, 0, stream>>>(x, rw, top_i, top_w);
    moe_cvt<<<(TOKENS * HID) / (256 * 4), 256, 0, stream>>>((const float4*)x, (u16x4*)xb);
    moe_compact<<<1, 256, 0, stream>>>(top_i, top_w, expert_off, tok_idx, ge_row);

    dim3 g1(INTER / 64, TOKENS / 64, NEXP);   // (32, 64, 8)
    moe_gemm1<<<g1, 256, 0, stream>>>(xb, w1, w1s, expert_off, tok_idx, a_bf);

    dim3 g2(HID / 64, TOKENS / 64, NEXP);     // (16, 64, 8)
    moe_gemm2<<<g2, 256, 0, stream>>>(a_bf, w2, w2s, expert_off, tok_idx, ge_row, out);
}
